// DilatedJumpLayer_56813827391734
// MI455X (gfx1250) — matrix operand order repaired and checked by reading, not compiled, not run on hardware
//
#include <hip/hip_runtime.h>
#include <stdint.h>

// ---------------- types ----------------
typedef __attribute__((ext_vector_type(16))) __bf16 v16bf;
typedef __attribute__((ext_vector_type(8)))  float  v8f;
typedef __attribute__((ext_vector_type(4)))  unsigned int u32x4;

union FragBF { v16bf v; u32x4 q[2]; };
union Pk128  { u32x4 q; unsigned short us[8]; };

__device__ __forceinline__ unsigned short f2bf(float f) {
  union { float f; uint32_t u; } v; v.f = f;
  uint32_t u = v.u;
  return (unsigned short)((u + 0x7FFFu + ((u >> 16) & 1u)) >> 16);
}

enum { EPI_LIN = 0, EPI_GELU = 1, EPI_SILU = 2, EPI_SIG2 = 3, EPI_PSI = 4 };

// ---------------- WMMA GEMM ----------------
// C[M,N] (+ epilogue) = A[M,K] * B[K,N]
// AMODE 0: A plain row-major, lda
// AMODE 1: A = activation [B*S, 512]; K = 3*512, tap = k/512, row = clamp(s+(tap-1)*dil)
// AMODE 2: A = circulant Cmat [1024,1024], row = m % 1024
// BLAYOUT 0: B stored n-major: B[n][k], ldb          (weights)
// BLAYOUT 1: B stored k-major: B[k][n], ld=512, per-batch base (seasonal activations)
// Block: 256 thr / 8 waves; tile 128x64, BK=64; wave tile 32x32 (2x2 WMMA);
// software pipeline: next global tile prefetched into VGPRs during WMMAs.
template<int AMODE, int BLAYOUT, int EPI, int CBF16>
__global__ __launch_bounds__(256)
void gemm_wmma(const unsigned short* __restrict__ A, long lda, int dil,
               const unsigned short* __restrict__ Bw, long ldb,
               const float* __restrict__ bias,
               float* __restrict__ Cf, unsigned short* __restrict__ Cb, long ldc,
               const float* __restrict__ psiK,
               int M, int N, int K) {
  __shared__ __attribute__((aligned(16))) unsigned short As[128 * 72];
  __shared__ __attribute__((aligned(16))) unsigned short Bs[64 * 72];

  const int tid = threadIdx.x;
  const int m0 = blockIdx.y * 128;
  const int n0 = blockIdx.x * 64;

  const unsigned short* Bp = Bw;
  if (AMODE == 2) Bp += ((size_t)(m0 >> 10)) << 19;  // per-batch x slab (1024*512)

  const int lane = tid & 31;
  const int wid  = tid >> 5;
  const int wm = wid & 3;        // 4 waves along M
  const int wn = wid >> 2;       // 2 waves along N
  const int lr = lane & 15;
  const int half = lane >> 4;

  v8f acc[2][2];
  const v8f vzero = {0.f,0.f,0.f,0.f,0.f,0.f,0.f,0.f};
#pragma unroll
  for (int i = 0; i < 2; i++)
#pragma unroll
    for (int j = 0; j < 2; j++) acc[i][j] = vzero;

  // staging maps
  const int arow = tid >> 1;            // A: row 0..127
  const int ac0  = (tid & 1) * 32;      // A: col 0 / 32  (4 x b128 each)
  const int bn_  = tid >> 2;            // B NK: n 0..63
  const int bc0  = (tid & 3) * 16;      // B NK: col (2 x b128)
  const int bkr  = tid >> 3;            // B KN: k 0..31 (+32)
  const int bnc  = (tid & 7) * 8;       // B KN: n 0..56

  u32x4 aR[4];
  Pk128 bR[2];

  // ---- prefetch helpers ----
  auto loadA = [&](int k0) {
#pragma unroll
    for (int j = 0; j < 4; j++) {
      int c = ac0 + j * 8;
      size_t src;
      if (AMODE == 0) {
        src = (size_t)(m0 + arow) * (size_t)lda + (size_t)(k0 + c);
      } else if (AMODE == 1) {
        int g = m0 + arow;
        int b = g >> 10, s = g & 1023;
        int tap = k0 >> 9;
        int kc  = (k0 & 511) + c;
        int ss = s + (tap - 1) * dil;
        ss = ss < 0 ? 0 : (ss > 1023 ? 1023 : ss);
        src = (((size_t)((b << 10) + ss)) << 9) + (size_t)kc;
      } else {
        src = (((size_t)((m0 & 1023) + arow)) << 10) + (size_t)(k0 + c);
      }
      aR[j] = *(const u32x4*)&A[src];
    }
  };
  auto loadB = [&](int k0) {
    if (BLAYOUT == 0) {
#pragma unroll
      for (int j = 0; j < 2; j++) {
        size_t src = (size_t)(n0 + bn_) * (size_t)ldb + (size_t)(k0 + bc0 + j * 8);
        bR[j].q = *(const u32x4*)&Bp[src];
      }
    } else {
#pragma unroll
      for (int j = 0; j < 2; j++) {
        size_t src = (((size_t)(k0 + bkr + j * 32)) << 9) + (size_t)(n0 + bnc);
        bR[j].q = *(const u32x4*)&Bp[src];
      }
    }
  };
  auto storeTiles = [&]() {
#pragma unroll
    for (int j = 0; j < 4; j++)
      *(u32x4*)&As[arow * 72 + ac0 + j * 8] = aR[j];
    if (BLAYOUT == 0) {
#pragma unroll
      for (int j = 0; j < 2; j++)
        *(u32x4*)&Bs[bn_ * 72 + bc0 + j * 8] = bR[j].q;
    } else {
#pragma unroll
      for (int j = 0; j < 2; j++)
#pragma unroll
        for (int e = 0; e < 8; e++)
          Bs[(bnc + e) * 72 + bkr + j * 32] = bR[j].us[e];
    }
  };

  loadA(0);
  loadB(0);

  for (int k0 = 0; k0 < K; k0 += 64) {
    storeTiles();
    __syncthreads();
    if (k0 + 64 < K) {          // prefetch next tile while WMMAs run
      loadA(k0 + 64);
      loadB(k0 + 64);
    }
#pragma unroll
    for (int ks = 0; ks < 2; ks++) {
      FragBF af[2], bf[2];
#pragma unroll
      for (int tm = 0; tm < 2; tm++) {
        int row = wm * 32 + tm * 16 + lr;
        af[tm].q[0] = *(const u32x4*)&As[row * 72 + ks * 32 + half * 8];       // K 0-7 / 8-15
        af[tm].q[1] = *(const u32x4*)&As[row * 72 + ks * 32 + 16 + half * 8];  // K 16-23 / 24-31
      }
#pragma unroll
      for (int tn = 0; tn < 2; tn++) {
        int col = wn * 32 + tn * 16 + lr;
        bf[tn].q[0] = *(const u32x4*)&Bs[col * 72 + ks * 32 + half * 8];       // K 0-7 / 8-15
        bf[tn].q[1] = *(const u32x4*)&Bs[col * 72 + ks * 32 + 16 + half * 8];  // K 16-23 / 24-31
      }
#pragma unroll
      for (int tm = 0; tm < 2; tm++)
#pragma unroll
        for (int tn = 0; tn < 2; tn++)
          acc[tm][tn] = __builtin_amdgcn_wmma_f32_16x16x32_bf16(
              false, af[tm].v, false, bf[tn].v, (short)0, acc[tm][tn], false, false);
    }
    __syncthreads();
  }

  // ---- epilogue ----
#pragma unroll
  for (int tm = 0; tm < 2; tm++) {
    int mb = m0 + wm * 32 + tm * 16 + half * 8;
#pragma unroll
    for (int tn = 0; tn < 2; tn++) {
      int n = n0 + wn * 32 + tn * 16 + lr;
      float bv = bias ? bias[n] : 0.0f;
#pragma unroll
      for (int i = 0; i < 8; i++) {
        int m = mb + i;
        float v = acc[tm][tn][i] + bv;
        if (EPI == EPI_GELU)      v = 0.5f * v * (1.0f + erff(v * 0.70710678118f));
        else if (EPI == EPI_SILU) v = v / (1.0f + __expf(-v));
        else if (EPI == EPI_SIG2) v = 2.0f / (1.0f + __expf(-v));
        size_t o = (size_t)m * (size_t)ldc + (size_t)n;
        if (EPI == EPI_PSI)  Cf[o] += psiK[m] * v;
        else if (CBF16)      Cb[o] = f2bf(v);
        else                 Cf[o] = v;
      }
    }
  }
}

// ---------------- small kernels ----------------
__global__ void k_cmat(unsigned short* __restrict__ C) {
  int e = blockIdx.x * 256 + threadIdx.x;
  int s = e >> 10, t = e & 1023;
  int m = s - t;
  float val;
  if (m == 0) val = 203.0f / 1024.0f;
  else {
    float th = 3.14159265358979323846f * (float)m / 1024.0f;
    val = sinf(203.0f * th) / (1024.0f * sinf(th));
  }
  C[e] = f2bf(val);
}

// [O][C][3] f32 -> [O][3*C] bf16, optional per-o scale = gamma[o]*rs
__global__ void k_pack3(const float* __restrict__ w, unsigned short* __restrict__ dst,
                        int O, int C, const float* __restrict__ gamma, float rs) {
  int idx = blockIdx.x * 256 + threadIdx.x;
  if (idx >= O * C) return;
  int o = idx / C, c = idx % C;
  float sc = gamma ? gamma[o] * rs : 1.0f;
  const float* s = w + (size_t)idx * 3;
  unsigned short* d = dst + (size_t)o * 3 * C + c;
  d[0]     = f2bf(s[0] * sc);
  d[C]     = f2bf(s[1] * sc);
  d[2 * C] = f2bf(s[2] * sc);
}

__global__ void k_cvt(const float* __restrict__ s, unsigned short* __restrict__ d, int n) {
  int i = blockIdx.x * 256 + threadIdx.x;
  if (i < n) d[i] = f2bf(s[i]);
}

__global__ void k_wavpack(const float* __restrict__ w, unsigned short* __restrict__ d) {
  int idx = blockIdx.x * 256 + threadIdx.x;  // 4*512*512
  int k = idx >> 18, rem = idx & 262143;
  int e = rem >> 9, dd = rem & 511;
  d[(k << 18) + (e << 9) + dd] = f2bf(w[(k << 18) + (dd << 9) + e]);
}

__global__ void k_hb1(const float* __restrict__ b, const float* __restrict__ gamma,
                      const float* __restrict__ beta, float* __restrict__ o, float rs) {
  int i = threadIdx.x;
  if (i < 512) o[i] = b[i] * gamma[i] * rs + beta[i];
}

__global__ void k_post(const float* __restrict__ sm, const float* __restrict__ x,
                       unsigned short* __restrict__ smB, unsigned short* __restrict__ dfB,
                       unsigned short* __restrict__ slB, int n, int flags) {
  int i = blockIdx.x * 256 + threadIdx.x;
  if (i >= n) return;
  float v = sm[i];
  smB[i] = f2bf(v);
  if (flags & 1) dfB[i] = f2bf(x[i] - v);
  if (flags & 2) slB[i] = f2bf(v / (1.0f + __expf(-v)));
}

__global__ void k_mean(const float* __restrict__ h, float* __restrict__ featP) {
  // grid (16,8), block 256; h is [16*1024, 256]
  int b = blockIdx.x, p = blockIdx.y, t = threadIdx.x;
  float acc = 0.f;
  for (int s = p * 128; s < p * 128 + 128; s++)
    acc += h[((size_t)(b * 1024 + s)) * 256 + t];
  featP[((p * 16 + b) * 256) + t] = acc;
}

__global__ void k_projdyn(const float* __restrict__ featP, const float* __restrict__ pw,
                          const float* __restrict__ pb, float* __restrict__ dynA,
                          float* __restrict__ dynB) {
  int t = threadIdx.x;            // 128 threads: b(4b) k(2b) j(1b)
  int b = t >> 3, k = (t >> 1) & 3, j = t & 1;
  int rowp = 2 * k + j;
  float acc = pb[rowp];
  for (int e = 0; e < 256; e++) {
    float f = 0.f;
    for (int p = 0; p < 8; p++) f += featP[(p * 16 + b) * 256 + e];
    acc += (f * (1.0f / 1024.0f)) * pw[rowp * 256 + e];
  }
  if (j == 0) {
    float sp = (acc > 20.f) ? acc : log1pf(__expf(acc));
    dynA[b * 4 + k] = sp + 0.01f;
  } else {
    dynB[b * 4 + k] = 1024.0f / (1.0f + __expf(-acc));
  }
}

__global__ void k_psi(const float* __restrict__ dynA, const float* __restrict__ dynB,
                      float* __restrict__ psiA) {
  int idx = blockIdx.x * 256 + threadIdx.x;  // 4*16384
  int k = idx >> 14, r = idx & 16383;
  int b = r >> 10, s = r & 1023;
  float u = ((float)s - dynB[b * 4 + k]) / dynA[b * 4 + k];
  psiA[idx] = 0.8673250705840776f * (1.0f - u * u) * __expf(-0.5f * u * u);
}

__global__ void k_zero(float* __restrict__ p, int n) {
  int i = blockIdx.x * 256 + threadIdx.x;
  if (i < n) p[i] = 0.f;
}

__global__ void k_final(const float* __restrict__ baseF, const float* __restrict__ kanF,
                        const float* __restrict__ ampF, const float* __restrict__ x,
                        const float* __restrict__ lg, const float* __restrict__ lb,
                        float* __restrict__ out) {
  __shared__ float red[256];
  int r = blockIdx.x, t = threadIdx.x;
  size_t o0 = (size_t)r * 512 + t, o1 = o0 + 256;
  float y0 = baseF[o0] + kanF[o0] * ampF[o0] + x[o0];
  float y1 = baseF[o1] + kanF[o1] * ampF[o1] + x[o1];
  red[t] = y0 + y1; __syncthreads();
  for (int st = 128; st > 0; st >>= 1) { if (t < st) red[t] += red[t + st]; __syncthreads(); }
  float mu = red[0] * (1.0f / 512.0f);
  __syncthreads();
  float d0 = y0 - mu, d1 = y1 - mu;
  red[t] = d0 * d0 + d1 * d1; __syncthreads();
  for (int st = 128; st > 0; st >>= 1) { if (t < st) red[t] += red[t + st]; __syncthreads(); }
  float rstd = rsqrtf(red[0] * (1.0f / 512.0f) + 1e-5f);
  out[o0] = d0 * rstd * lg[t] + lb[t];
  out[o1] = d1 * rstd * lg[t + 256] + lb[t + 256];
}

// ---------------- host ----------------
extern "C" void kernel_launch(void* const* d_in, const int* in_sizes, int n_in,
                              void* d_out, int out_size, void* d_ws, size_t ws_size,
                              hipStream_t stream) {
  (void)in_sizes; (void)n_in; (void)out_size; (void)ws_size;
  const float* x        = (const float*)d_in[0];
  const float* conv_w   = (const float*)d_in[1];
  const float* conv_b   = (const float*)d_in[2];
  const float* fusion_w = (const float*)d_in[3];
  const float* fusion_b = (const float*)d_in[4];
  const float* refine_w = (const float*)d_in[5];
  const float* refine_b = (const float*)d_in[6];
  const float* hyp1_w   = (const float*)d_in[7];
  const float* hyp1_b   = (const float*)d_in[8];
  const float* bn_gamma = (const float*)d_in[9];
  const float* bn_beta  = (const float*)d_in[10];
  const float* hyp2_w   = (const float*)d_in[11];
  const float* hyp2_b   = (const float*)d_in[12];
  const float* proj_w   = (const float*)d_in[13];
  const float* proj_b   = (const float*)d_in[14];
  const float* gate_w   = (const float*)d_in[15];
  const float* gate_b   = (const float*)d_in[16];
  const float* base_w   = (const float*)d_in[17];
  const float* base_b   = (const float*)d_in[18];
  const float* wav_w    = (const float*)d_in[19];
  const float* wav_b    = (const float*)d_in[20];
  const float* ln_g     = (const float*)d_in[21];
  const float* ln_b     = (const float*)d_in[22];
  float* out = (float*)d_out;

  const int BS = 16384;          // 16*1024
  const int NE = BS * 512;       // 8388608 elements per [BS,512]
  const float rsbn = rsqrtf(1.0f + 1e-5f);

  char* wp = (char*)d_ws;
  auto alloc = [&](size_t bytes) -> char* {
    char* p = wp; wp += (bytes + 255) & ~((size_t)255); return p;
  };
  unsigned short* Cmat   = (unsigned short*)alloc(1024ull * 1024 * 2);
  unsigned short* Wc     = (unsigned short*)alloc(4ull * 512 * 1536 * 2);
  unsigned short* Wref   = (unsigned short*)alloc(512ull * 1536 * 2);
  unsigned short* Wh1    = (unsigned short*)alloc(512ull * 1536 * 2);
  unsigned short* Wh2    = (unsigned short*)alloc(256ull * 1536 * 2);
  unsigned short* Wfus   = (unsigned short*)alloc(512ull * 2048 * 2);
  unsigned short* Wbase  = (unsigned short*)alloc(512ull * 512 * 2);
  unsigned short* Wgate  = (unsigned short*)alloc(512ull * 512 * 2);
  unsigned short* Wwav   = (unsigned short*)alloc(4ull * 512 * 512 * 2);
  float*          hb1    = (float*)alloc(512 * 4);
  unsigned short* xbf    = (unsigned short*)alloc((size_t)NE * 2);
  float*          smoothF= (float*)alloc((size_t)NE * 4);
  unsigned short* smoothB= (unsigned short*)alloc((size_t)NE * 2);
  unsigned short* diffB  = (unsigned short*)alloc((size_t)NE * 2);
  unsigned short* spikeB = (unsigned short*)alloc((size_t)NE * 2);
  unsigned short* siluB  = (unsigned short*)alloc((size_t)NE * 2);
  unsigned short* outsB  = (unsigned short*)alloc((size_t)BS * 2048 * 2); // 67MB, reused below
  float*          featP  = (float*)alloc(8ull * 16 * 256 * 4);
  float*          dynA   = (float*)alloc(16 * 4 * 4);
  float*          dynB   = (float*)alloc(16 * 4 * 4);
  float*          psiA   = (float*)alloc(4ull * BS * 4);
  float*          baseF  = (float*)alloc((size_t)NE * 4);
  float*          kanF   = (float*)alloc((size_t)NE * 4);
  float*          ampF   = (float*)alloc((size_t)NE * 4);
  // aliases inside the dead outsB region (only used after fusion iter2 is done)
  unsigned short* h1B   = outsB;
  float*          h2F   = (float*)((char*)outsB + (((size_t)NE * 2 + 255) & ~(size_t)255));
  unsigned short* baseB = (unsigned short*)((char*)h2F + (((size_t)BS * 256 * 4 + 255) & ~(size_t)255));

  // ---- prep / packing ----
  k_cmat<<<4096, 256, 0, stream>>>(Cmat);
  for (int i = 0; i < 4; i++)
    k_pack3<<<1024, 256, 0, stream>>>(conv_w + (size_t)i * 512 * 512 * 3,
                                      Wc + (size_t)i * 512 * 1536, 512, 512, nullptr, 1.f);
  k_pack3<<<1024, 256, 0, stream>>>(refine_w, Wref, 512, 512, nullptr, 1.f);
  k_pack3<<<1024, 256, 0, stream>>>(hyp1_w, Wh1, 512, 512, bn_gamma, rsbn);
  k_pack3<<<512, 256, 0, stream>>>(hyp2_w, Wh2, 256, 512, nullptr, 1.f);
  k_cvt<<<4096, 256, 0, stream>>>(fusion_w, Wfus, 512 * 2048);
  k_cvt<<<1024, 256, 0, stream>>>(base_w, Wbase, 512 * 512);
  k_cvt<<<1024, 256, 0, stream>>>(gate_w, Wgate, 512 * 512);
  k_wavpack<<<4096, 256, 0, stream>>>(wav_w, Wwav);
  k_hb1<<<1, 512, 0, stream>>>(hyp1_b, bn_gamma, bn_beta, hb1, rsbn);
  k_cvt<<<NE / 256, 256, 0, stream>>>(x, xbf, NE);

  dim3 g512(512 / 64, BS / 128), g256(256 / 64, BS / 128);

  // ---- seasonal low-pass: smoothF = Dirichlet-circulant @ x ----
  gemm_wmma<2, 1, EPI_LIN, 0><<<g512, 256, 0, stream>>>(
      Cmat, 1024, 0, xbf, 512, nullptr, smoothF, nullptr, 512, nullptr, BS, 512, 1024);
  k_post<<<NE / 256, 256, 0, stream>>>(smoothF, x, smoothB, diffB, siluB, NE, 0);

  // ---- 2 smoothing iterations ----
  const int DIL[4] = {1, 2, 4, 8};
  for (int it = 0; it < 2; it++) {
    for (int i = 0; i < 4; i++)
      gemm_wmma<1, 0, EPI_GELU, 1><<<g512, 256, 0, stream>>>(
          smoothB, 512, DIL[i], Wc + (size_t)i * 512 * 1536, 1536,
          conv_b + i * 512, nullptr, outsB + i * 512, 2048, nullptr, BS, 512, 1536);
    gemm_wmma<0, 0, EPI_LIN, 0><<<g512, 256, 0, stream>>>(
        outsB, 2048, 0, Wfus, 2048, fusion_b, smoothF, nullptr, 512, nullptr, BS, 512, 2048);
    k_post<<<NE / 256, 256, 0, stream>>>(smoothF, x, smoothB, diffB, siluB, NE,
                                         it == 1 ? 3 : 0);
  }
  // refine only matters after the last iteration (iter-1 spike is dead)
  gemm_wmma<1, 0, EPI_LIN, 1><<<g512, 256, 0, stream>>>(
      diffB, 512, 1, Wref, 1536, refine_b, nullptr, spikeB, 512, nullptr, BS, 512, 1536);

  // ---- hypernet ----
  gemm_wmma<1, 0, EPI_SILU, 1><<<g512, 256, 0, stream>>>(
      smoothB, 512, 1, Wh1, 1536, hb1, nullptr, h1B, 512, nullptr, BS, 512, 1536);
  gemm_wmma<1, 0, EPI_SILU, 0><<<g256, 256, 0, stream>>>(
      h1B, 512, 1, Wh2, 1536, hyp2_b, h2F, nullptr, 256, nullptr, BS, 256, 1536);
  k_mean<<<dim3(16, 8), 256, 0, stream>>>(h2F, featP);
  k_projdyn<<<1, 128, 0, stream>>>(featP, proj_w, proj_b, dynA, dynB);
  k_psi<<<256, 256, 0, stream>>>(dynA, dynB, psiA);

  // ---- base path + gate + wavelet-KAN ----
  gemm_wmma<0, 0, EPI_LIN, 0><<<g512, 256, 0, stream>>>(
      siluB, 512, 0, Wbase, 512, base_b, baseF, nullptr, 512, nullptr, BS, 512, 512);
  k_cvt<<<NE / 256, 256, 0, stream>>>(baseF, baseB, NE);
  gemm_wmma<0, 0, EPI_SIG2, 0><<<g512, 256, 0, stream>>>(
      baseB, 512, 0, Wgate, 512, gate_b, ampF, nullptr, 512, nullptr, BS, 512, 512);
  k_zero<<<NE / 256, 256, 0, stream>>>(kanF, NE);
  for (int k = 0; k < 4; k++)
    gemm_wmma<0, 0, EPI_PSI, 0><<<g512, 256, 0, stream>>>(
        spikeB, 512, 0, Wwav + (size_t)k * 512 * 512, 512, wav_b + k * 512,
        kanF, nullptr, 512, psiA + (size_t)k * BS, BS, 512, 512);

  // ---- residual + LayerNorm ----
  k_final<<<BS, 256, 0, stream>>>(baseF, kanF, ampF, x, ln_g, ln_b, out);
}
